// OFNAttentionBranch_7997229105588
// MI455X (gfx1250) — matrix operand order repaired and checked by reading, not compiled, not run on hardware
//
#include <hip/hip_runtime.h>
#include <hip/hip_bf16.h>

typedef _Float16 h16;
typedef __attribute__((ext_vector_type(16))) _Float16 v16h;
typedef __attribute__((ext_vector_type(8)))  _Float16 v8h;
typedef __attribute__((ext_vector_type(8)))  float    v8f;

static constexpr int SB  = 2;     // batch
static constexpr int SS  = 2048;  // sequence
static constexpr int SD  = 1024;  // model dim
static constexpr int SH  = 16;    // heads
static constexpr int SDH = 64;    // head dim
static constexpr int SWIN = 256;  // window

#define WMMA_F16(a, b, c) \
  __builtin_amdgcn_wmma_f32_16x16x32_f16(false, (a), false, (b), (short)0, (c), false, false)

// ---------------------------------------------------------------- conversions
__global__ void convert_x_f16(const float* __restrict__ x, h16* __restrict__ x16, int n) {
  int i = blockIdx.x * blockDim.x + threadIdx.x;
  if (i < n) x16[i] = (h16)x[i];
}

// W row-major [K=1024][N=1024] fp32 -> Wt f16 [N][K]
__global__ void transpose_w_f16(const float* __restrict__ W, h16* __restrict__ Wt) {
  int idx = blockIdx.x * blockDim.x + threadIdx.x;   // over 1M
  int nn = idx >> 10;
  int k  = idx & 1023;
  Wt[(size_t)nn * SD + k] = (h16)W[(size_t)k * SD + nn];
}

// --------------------------------------------------------------------------
// Block-level GEMM: 256 threads = 8 waves; block computes rows
// [row0_blk, row0_blk+256) x cols [col0, col0+64). Wave w owns 32 rows
// (two 16-row A fragments), so each LDS B fragment feeds two WMMAs.
// Weight slice (64 cols x 32 k, f16, 4KB) is staged into LDS with
// global_load_async_to_lds_b128 (1 x 16B per thread), double-buffered and
// tracked with ASYNCcnt; WMMA consumes B fragments from LDS while the next
// chunk streams in.
// --------------------------------------------------------------------------
__device__ __forceinline__ void block_gemm_asyncB(
    const h16* __restrict__ A, const h16* __restrict__ Bt,
    int row0_wave, int col0, int K, h16* Bs /* [2][64*32] */, v8f acc[2][4]) {
  const int t    = threadIdx.x;
  const int lane = t & 31;
  const int m    = lane & 15;
  const int hi   = lane >> 4;
  const h16* arow0 = A + (size_t)(row0_wave + m) * K;
  const h16* arow1 = A + (size_t)(row0_wave + 16 + m) * K;

  // async-copy assignment: thread t moves 8 halves of column c, segment seg
  const int c   = t >> 2;   // 0..63 local column
  const int seg = t & 3;    // 0..3  (8-half segment within 32-k chunk)
  const h16* gsrc = Bt + (size_t)(col0 + c) * K + seg * 8;
  const uint32_t bsbase = (uint32_t)(size_t)Bs;          // LDS byte offset
  const uint32_t dstoff = (uint32_t)((c * 32 + seg * 8) * sizeof(h16));

  // preload chunk 0 into buffer 0
  {
    uint32_t ldsb = bsbase + dstoff;
    unsigned long long ga = (unsigned long long)(size_t)gsrc;
    asm volatile("global_load_async_to_lds_b128 %0, %1, off"
                 :: "v"(ldsb), "v"(ga) : "memory");
  }

  for (int kb = 0; kb < K; kb += 32) {
    // current chunk resident for the whole block
    asm volatile("s_wait_asynccnt 0x0" ::: "memory");
    __syncthreads();
    const int buf = (kb >> 5) & 1;

    // kick off next chunk into the other buffer
    if (kb + 32 < K) {
      uint32_t ldsb = bsbase + (uint32_t)((buf ^ 1) * 64 * 32 * sizeof(h16)) + dstoff;
      unsigned long long ga = (unsigned long long)(size_t)(gsrc + kb + 32);
      asm volatile("global_load_async_to_lds_b128 %0, %1, off"
                   :: "v"(ldsb), "v"(ga) : "memory");
    }

    // two A fragments (wave-private, direct from global)
    v16h a0, a1;
    {
      v8h lo0 = *(const v8h*)(arow0 + kb + hi * 16);
      v8h hi0 = *(const v8h*)(arow0 + kb + hi * 16 + 8);
      v8h lo1 = *(const v8h*)(arow1 + kb + hi * 16);
      v8h hi1 = *(const v8h*)(arow1 + kb + hi * 16 + 8);
#pragma unroll
      for (int i = 0; i < 8; ++i) {
        a0[i] = lo0[i]; a0[i + 8] = hi0[i];
        a1[i] = lo1[i]; a1[i + 8] = hi1[i];
      }
    }

    // B fragments from LDS; each one feeds two WMMAs (row-group 0 and 1),
    // so the second WMMA hides the next fragment's ds_load latency.
    const h16* bs = Bs + buf * (64 * 32);
    v16h bb[4];
#pragma unroll
    for (int j = 0; j < 4; ++j)
      bb[j] = *(const v16h*)(bs + (j * 16 + m) * 32 + hi * 16);
#pragma unroll
    for (int j = 0; j < 4; ++j) {
      acc[0][j] = WMMA_F16(a0, bb[j], acc[0][j]);
      acc[1][j] = WMMA_F16(a1, bb[j], acc[1][j]);
    }

    __syncthreads();   // everyone done reading buf before it gets overwritten
  }
}

// ------------------------------------------------------------- fused QKV GEMM
// grid.x: 256 blocks = 16 row-blocks (256 rows) x 16 col-blocks (64 cols);
// grid.y selects q/k/v.
__global__ void qkv_kernel(const h16* __restrict__ x16,
                           const h16* __restrict__ Wqt, const h16* __restrict__ Wkt,
                           const h16* __restrict__ Wvt,
                           const float* __restrict__ bq, const float* __restrict__ bk,
                           const float* __restrict__ bv,
                           h16* __restrict__ q16, h16* __restrict__ k16,
                           h16* __restrict__ v16t) {
  __shared__ __align__(16) h16 Bs[2][64 * 32];
  const int wid = threadIdx.x >> 5;
  const int rb = blockIdx.x >> 4;     // 0..15
  const int cb = blockIdx.x & 15;     // 0..15
  const int row0 = rb * 256 + wid * 32;
  const int col0 = cb * 64;
  const int which = blockIdx.y;
  const h16*   Wt   = (which == 0) ? Wqt : (which == 1) ? Wkt : Wvt;
  const float* bias = (which == 0) ? bq  : (which == 1) ? bk  : bv;

  v8f acc[2][4];
#pragma unroll
  for (int g = 0; g < 2; ++g)
#pragma unroll
    for (int j = 0; j < 4; ++j)
#pragma unroll
      for (int i = 0; i < 8; ++i) acc[g][j][i] = 0.0f;

  block_gemm_asyncB(x16, Wt, row0, col0, SD, &Bs[0][0], acc);

  const int lane = threadIdx.x & 31;
  const int n  = lane & 15;
  const int hi = lane >> 4;
#pragma unroll
  for (int j = 0; j < 4; ++j) {
    int gn = col0 + j * 16 + n;
    int hh = gn >> 6, d = gn & 63;
    float bb = bias[gn];
#pragma unroll
    for (int g = 0; g < 2; ++g) {
#pragma unroll
      for (int r = 0; r < 8; ++r) {
        int grow = row0 + g * 16 + hi * 8 + r;       // b*S + s
        int b = grow >> 11, s = grow & (SS - 1);
        float val = acc[g][j][r] + bb;
        if (which == 0)
          q16[((size_t)(b * SH + hh) * SS + s) * SDH + d] = (h16)(val * 0.125f);
        else if (which == 1)
          k16[((size_t)(b * SH + hh) * SS + s) * SDH + d] = (h16)val;
        else
          v16t[((size_t)(b * SH + hh) * SDH + d) * SS + s] = (h16)val;
      }
    }
  }
}

// ------------------------------------------- sliding-window flash attention
// One wave per (b, h, 16-query tile). Keys processed 32 at a time.
__global__ void attn_kernel(const h16* __restrict__ q16, const h16* __restrict__ k16,
                            const h16* __restrict__ v16t, h16* __restrict__ ctx16) {
  __shared__ __align__(16) h16 Plds[8][16 * 32];   // per-wave P staging
  const int wid  = threadIdx.x >> 5;
  const int wave = blockIdx.x * (blockDim.x >> 5) + wid;
  const int qt = wave & 127;                  // query tile
  const int bh = wave >> 7;                   // b*16 + h
  const int q0 = qt * 16;
  const int lane = threadIdx.x & 31;
  const int n  = lane & 15;
  const int hi = lane >> 4;

  const h16* qb = q16  + (size_t)bh * SS * SDH;
  const h16* kb = k16  + (size_t)bh * SS * SDH;
  const h16* vb = v16t + (size_t)bh * SDH * SS;

  // Q fragments (A-matrix, rows = queries), resident for whole loop
  const h16* qrow = qb + (size_t)(q0 + n) * SDH;
  v16h aq0, aq1;
  {
    v8h l0 = *(const v8h*)(qrow + hi * 16);
    v8h l1 = *(const v8h*)(qrow + hi * 16 + 8);
    v8h l2 = *(const v8h*)(qrow + 32 + hi * 16);
    v8h l3 = *(const v8h*)(qrow + 40 + hi * 16);
#pragma unroll
    for (int i = 0; i < 8; ++i) {
      aq0[i] = l0[i]; aq0[i + 8] = l1[i];
      aq1[i] = l2[i]; aq1[i + 8] = l3[i];
    }
  }

  v8f o[4];
  float rmax[8], rsum[8];
#pragma unroll
  for (int j = 0; j < 4; ++j)
#pragma unroll
    for (int i = 0; i < 8; ++i) o[j][i] = 0.0f;
#pragma unroll
  for (int r = 0; r < 8; ++r) { rmax[r] = -1e30f; rsum[r] = 0.0f; }

  const int jbeg = (q0 >= 240) ? ((q0 - 240) & ~31) : 0;
  for (int j0 = jbeg; j0 < q0 + 16; j0 += 32) {
    // ---- scores S = Q.K^T (already scaled via q), 16x32 per block
    int kr0 = j0 + n;        if (kr0 > SS - 1) kr0 = SS - 1;
    int kr1 = j0 + 16 + n;   if (kr1 > SS - 1) kr1 = SS - 1;
    const h16* krow0 = kb + (size_t)kr0 * SDH;
    const h16* krow1 = kb + (size_t)kr1 * SDH;
    v16h b00 = *(const v16h*)(krow0 + hi * 16);
    v16h b01 = *(const v16h*)(krow0 + 32 + hi * 16);
    v16h b10 = *(const v16h*)(krow1 + hi * 16);
    v16h b11 = *(const v16h*)(krow1 + 32 + hi * 16);
    v8f s0, s1;
#pragma unroll
    for (int i = 0; i < 8; ++i) { s0[i] = 0.0f; s1[i] = 0.0f; }
    s0 = WMMA_F16(aq0, b00, s0);
    s0 = WMMA_F16(aq1, b01, s0);
    s1 = WMMA_F16(aq0, b10, s1);
    s1 = WMMA_F16(aq1, b11, s1);

    // ---- online softmax (rows striped: lane half holds rows hi*8+r, col n)
    float corr[8], p0[8], p1[8];
#pragma unroll
    for (int r = 0; r < 8; ++r) {
      int qi  = q0 + hi * 8 + r;
      int kj0 = j0 + n;
      int kj1 = kj0 + 16;
      float sv0 = s0[r], sv1 = s1[r];
      int d0 = qi - kj0, d1 = qi - kj1;
      if (!(d0 >= 0 && d0 < SWIN)) sv0 = -1e30f;
      if (!(d1 >= 0 && d1 < SWIN)) sv1 = -1e30f;
      float loc = fmaxf(sv0, sv1);
      loc = fmaxf(loc, __shfl_xor(loc, 1, 32));
      loc = fmaxf(loc, __shfl_xor(loc, 2, 32));
      loc = fmaxf(loc, __shfl_xor(loc, 4, 32));
      loc = fmaxf(loc, __shfl_xor(loc, 8, 32));
      float nm = fmaxf(rmax[r], loc);
      corr[r] = __expf(rmax[r] - nm);
      p0[r] = __expf(sv0 - nm);
      p1[r] = __expf(sv1 - nm);
      float ts = p0[r] + p1[r];
      ts += __shfl_xor(ts, 1, 32);
      ts += __shfl_xor(ts, 2, 32);
      ts += __shfl_xor(ts, 4, 32);
      ts += __shfl_xor(ts, 8, 32);
      rsum[r] = rsum[r] * corr[r] + ts;
      rmax[r] = nm;
    }
#pragma unroll
    for (int j = 0; j < 4; ++j)
#pragma unroll
      for (int r = 0; r < 8; ++r) o[j][r] *= corr[r];

    // ---- relayout P (C-layout -> A-layout) through wave-private LDS
    h16* pl = &Plds[wid][0];
#pragma unroll
    for (int r = 0; r < 8; ++r) {
      pl[(hi * 8 + r) * 32 + n]      = (h16)p0[r];
      pl[(hi * 8 + r) * 32 + 16 + n] = (h16)p1[r];
    }
    v16h ap;
    {
      const h16* prow = pl + n * 32;
      v8h l0 = *(const v8h*)(prow + hi * 16);
      v8h l1 = *(const v8h*)(prow + hi * 16 + 8);
#pragma unroll
      for (int i = 0; i < 8; ++i) { ap[i] = l0[i]; ap[i + 8] = l1[i]; }
    }

    // ---- O += P.V  (V transposed [d][s]: contiguous key runs)
    int jv = j0 + hi * 16; if (jv > SS - 16) jv = SS - 16;
    v16h bvf[4];
#pragma unroll
    for (int jb = 0; jb < 4; ++jb) {
      const h16* vrow = vb + (size_t)(jb * 16 + n) * SS;
      bvf[jb] = *(const v16h*)(vrow + jv);
    }
#pragma unroll
    for (int jb = 0; jb < 4; ++jb)
      o[jb] = WMMA_F16(ap, bvf[jb], o[jb]);
  }

  // ---- normalize and emit context rows as f16 [B*S, 1024]
  const int b = bh >> 4, hh = bh & 15;
#pragma unroll
  for (int r = 0; r < 8; ++r) {
    float inv = 1.0f / rsum[r];
    int s = q0 + hi * 8 + r;
#pragma unroll
    for (int jb = 0; jb < 4; ++jb) {
      int col = hh * 64 + jb * 16 + n;
      ctx16[(size_t)(b * SS + s) * SD + col] = (h16)(o[jb][r] * inv);
    }
  }
}

// -------------------------------------------------------- output projection
__global__ void oproj_kernel(const h16* __restrict__ ctx16, const h16* __restrict__ Wot,
                             const float* __restrict__ bo,
                             const unsigned char* __restrict__ kpm,
                             float* __restrict__ out) {
  __shared__ __align__(16) h16 Bs[2][64 * 32];
  const int wid = threadIdx.x >> 5;
  const int rb = blockIdx.x >> 4;
  const int cb = blockIdx.x & 15;
  const int row0 = rb * 256 + wid * 32;
  const int col0 = cb * 64;

  v8f acc[2][4];
#pragma unroll
  for (int g = 0; g < 2; ++g)
#pragma unroll
    for (int j = 0; j < 4; ++j)
#pragma unroll
      for (int i = 0; i < 8; ++i) acc[g][j][i] = 0.0f;

  block_gemm_asyncB(ctx16, Wot, row0, col0, SD, &Bs[0][0], acc);

  const int lane = threadIdx.x & 31;
  const int n  = lane & 15;
  const int hi = lane >> 4;
#pragma unroll
  for (int j = 0; j < 4; ++j) {
    int gn = col0 + j * 16 + n;
    float bb = bo[gn];
#pragma unroll
    for (int g = 0; g < 2; ++g) {
#pragma unroll
      for (int r = 0; r < 8; ++r) {
        int grow = row0 + g * 16 + hi * 8 + r;
        int b = grow >> 11, s = grow & (SS - 1);
        float mval = kpm[b * SS + s] ? 1.0f : 0.0f;
        out[(size_t)grow * SD + gn] = (acc[g][j][r] + bb) * mval;
      }
    }
  }
}

// ---------------------------------------------------------------------------
extern "C" void kernel_launch(void* const* d_in, const int* in_sizes, int n_in,
                              void* d_out, int out_size, void* d_ws, size_t ws_size,
                              hipStream_t stream) {
  const float* x  = (const float*)d_in[0];
  const unsigned char* kpm = (const unsigned char*)d_in[1];
  const float* Wq = (const float*)d_in[2];
  const float* bq = (const float*)d_in[3];
  const float* Wk = (const float*)d_in[4];
  const float* bk = (const float*)d_in[5];
  const float* Wv = (const float*)d_in[6];
  const float* bv = (const float*)d_in[7];
  const float* Wo = (const float*)d_in[8];
  const float* bo = (const float*)d_in[9];
  float* out = (float*)d_out;

  // workspace carve-up (f16 elements), ~48 MB total
  char* ws = (char*)d_ws;
  const size_t xN = (size_t)SB * SS * SD;       // 4M
  const size_t wN = (size_t)SD * SD;            // 1M
  h16* x16  = (h16*)ws;            ws += xN * sizeof(h16);
  h16* Wqt  = (h16*)ws;            ws += wN * sizeof(h16);
  h16* Wkt  = (h16*)ws;            ws += wN * sizeof(h16);
  h16* Wvt  = (h16*)ws;            ws += wN * sizeof(h16);
  h16* Wot  = (h16*)ws;            ws += wN * sizeof(h16);
  h16* q16  = (h16*)ws;            ws += xN * sizeof(h16);
  h16* k16  = (h16*)ws;            ws += xN * sizeof(h16);
  h16* v16t = (h16*)ws;            ws += xN * sizeof(h16);
  h16* ctx16 = (h16*)ws;           ws += xN * sizeof(h16);

  // 1) fp32 -> f16 conversions (x straight, weights transposed [N][K])
  convert_x_f16<<<(int)((xN + 255) / 256), 256, 0, stream>>>(x, x16, (int)xN);
  transpose_w_f16<<<(int)(wN / 256), 256, 0, stream>>>(Wq, Wqt);
  transpose_w_f16<<<(int)(wN / 256), 256, 0, stream>>>(Wk, Wkt);
  transpose_w_f16<<<(int)(wN / 256), 256, 0, stream>>>(Wv, Wvt);
  transpose_w_f16<<<(int)(wN / 256), 256, 0, stream>>>(Wo, Wot);

  // 2) fused QKV projections: 256 blocks (16 row-blk x 16 col-blk) x 3
  qkv_kernel<<<dim3(256, 3), 256, 0, stream>>>(x16, Wqt, Wkt, Wvt, bq, bk, bv,
                                               q16, k16, v16t);

  // 3) sliding-window flash attention: B*H*(S/16) = 4096 waves
  attn_kernel<<<512, 256, 0, stream>>>(q16, k16, v16t, ctx16);

  // 4) output projection + bias + query padding mask
  oproj_kernel<<<256, 256, 0, stream>>>(ctx16, Wot, bo, kpm, out);
}
